// MemristorDense_13013750907294
// MI455X (gfx1250) — matrix operand order repaired and checked by reading, not compiled, not run on hardware
//
#include <hip/hip_runtime.h>
#include <hip/hip_bf16.h>
#include <math.h>

// ---------------------------------------------------------------------------
// Memristive dense layer, reformulated as a stacked-K f16 WMMA GEMM.
//   I[b,j] = sum_i s*exp(mu*E) * Gv * (1 + s*z*E sigma + (sigma z E)^2/2)
//          = [A0|A1|A2](b, 3K) x [W0;W1;W2](3K, j)
// Workspace: [0..4)     max-weight bits
//            [256..)    A  f16 (512 x 2400)   = 2,457,600 B
//            [+A ..)    WT f16 (256 x 2400)   = 1,228,800 B   (~3.7 MB total)
// ---------------------------------------------------------------------------

typedef __attribute__((ext_vector_type(16))) _Float16 v16h;
typedef __attribute__((ext_vector_type(8)))  _Float16 v8h;
typedef __attribute__((ext_vector_type(8)))  float    v8f;

#define NIN   784
#define NOUT  128
#define BATCH 512
#define KDIM  (NIN + 1)        // 785 (inputs + bias row)
#define KP    800              // padded K per Taylor term (multiple of 32)
#define KP3   (3 * KP)         // stacked K = 2400
#define NCOL  (2 * NOUT)       // 256 interleaved pos/neg columns

#define V_REF   0.25f
#define K_V     0.5f
#define G_MIN   (1.0f / 1003.0f)
#define G_MAX   (1.0f / 284.6f)
#define D2D_STD 0.25f
// lognormal params for n (mean 2.132, std 0.095)
#define C0_EXP2 1.0907760f     // mu / ln2,  mu = ln(N^2/sqrt(N^2+S^2))
#define SIGMA_N 0.0445369f     // sqrt(ln(1 + S^2/N^2))

__device__ __forceinline__ unsigned hash_u32(unsigned x) {
    x ^= x >> 16; x *= 0x7feb352du;
    x ^= x >> 15; x *= 0x846ca68bu;
    x ^= x >> 16; return x;
}

// deterministic standard normal via hash + Box-Muller
__device__ __forceinline__ float gauss(unsigned idx, unsigned salt) {
    unsigned h1 = hash_u32(idx * 2654435761u + salt * 0x9e3779b9u + 0x85ebca6bu);
    unsigned h2 = hash_u32(h1 ^ (idx + 0xc2b2ae35u));
    float u1 = (float)((h1 >> 8) + 1u) * (1.0f / 16777216.0f);   // (0,1]
    float u2 = (float)(h2 >> 8) * (1.0f / 16777216.0f);
    return sqrtf(-2.0f * logf(u1)) * cosf(6.28318530717958f * u2);
}

// ---- kernel 1: seed global max with bias value 0.5 -------------------------
__global__ void k_init(unsigned* maxbits) {
    if (threadIdx.x == 0 && blockIdx.x == 0)
        *maxbits = __float_as_uint(0.5f);
}

// ---- kernel 2: max over w_pos/w_neg (all >= 0, so uint order == float) -----
__global__ __launch_bounds__(256) void k_maxw(const float* __restrict__ wp,
                                              const float* __restrict__ wn,
                                              unsigned* maxbits) {
    __shared__ float smax[256];
    const int tid = threadIdx.x;
    const int stride = gridDim.x * blockDim.x;
    float m = 0.0f;
    for (int i = blockIdx.x * blockDim.x + tid; i < NIN * NOUT; i += stride) {
        m = fmaxf(m, wp[i]);
        m = fmaxf(m, wn[i]);
    }
    smax[tid] = m;
    __syncthreads();
    for (int s = 128; s > 0; s >>= 1) {
        if (tid < s) smax[tid] = fmaxf(smax[tid], smax[tid + s]);
        __syncthreads();
    }
    if (tid == 0) atomicMax(maxbits, __float_as_uint(smax[0]));
}

// ---- kernel 3: device-side matrices W0/W1/W2, stored transposed ------------
// WT[j][t*KP + i], j = interleaved pos/neg column, i = input row.
__global__ __launch_bounds__(256) void k_build_wt(const float* __restrict__ wp,
                                                  const float* __restrict__ wn,
                                                  const float* __restrict__ bp,
                                                  const float* __restrict__ bn,
                                                  const unsigned* __restrict__ maxbits,
                                                  _Float16* __restrict__ WT) {
    const int i = blockIdx.x * blockDim.x + threadIdx.x;   // 0..KP
    const int j = blockIdx.y;                              // 0..255
    if (i >= KP) return;
    _Float16 w0 = (_Float16)0.0f, w1 = (_Float16)0.0f, w2 = (_Float16)0.0f;
    if (i < KDIM) {
        const int  oc  = j >> 1;
        const bool neg = (j & 1) != 0;
        const float w = (i < NIN) ? (neg ? wn[(size_t)i * NOUT + oc]
                                         : wp[(size_t)i * NOUT + oc])
                                  : (neg ? bn[oc] : bp[oc]);
        const float maxw = __uint_as_float(*maxbits);
        const float kG = (G_MAX - G_MIN) / maxw;
        float G = kG * w + G_MIN;
        const unsigned cell = (unsigned)(j * KDIM + i);
        G *= expf(-D2D_STD * gauss(cell, 1u));           // D2D lognormal on R
        const float zn = gauss(cell, 2u);                // n-parameter draw
        const float Gv = G * V_REF;
        w0 = (_Float16)Gv;
        w1 = (_Float16)(Gv * (SIGMA_N * zn));
        w2 = (_Float16)(Gv * (0.5f * SIGMA_N * SIGMA_N * zn * zn));
    }
    _Float16* row = WT + (size_t)j * KP3;
    row[i]          = w0;
    row[KP + i]     = w1;
    row[2 * KP + i] = w2;
}

// ---- kernel 4: activation-side matrices A0/A1/A2 ---------------------------
__global__ __launch_bounds__(256) void k_build_a(const float* __restrict__ x,
                                                 _Float16* __restrict__ A) {
    const int i = blockIdx.x * blockDim.x + threadIdx.x;   // 0..KP
    const int b = blockIdx.y;                              // 0..511
    if (i >= KP) return;
    _Float16 a0 = (_Float16)0.0f, a1 = (_Float16)0.0f, a2 = (_Float16)0.0f;
    if (i < KDIM) {
        const float xv = (i < NIN) ? x[(size_t)b * NIN + i] : 1.0f;
        const float t = 2.0f * fabsf(xv);                  // |V|/V_REF
        if (t > 0.0f) {
            const float E = log2f(t);
            const float s = (xv < 0.0f) ? -1.0f : 1.0f;
            const float A0 = s * exp2f(C0_EXP2 * E);       // s * exp(mu*E)
            a0 = (_Float16)A0;
            a1 = (_Float16)(A0 * E);
            a2 = (_Float16)(A0 * E * E);
        }
    }
    _Float16* row = A + (size_t)b * KP3;
    row[i]          = a0;
    row[KP + i]     = a1;
    row[2 * KP + i] = a2;
}

// ---- kernel 5: WMMA GEMM + fused differential-pair epilogue ----------------
// block = (32,8): 8 waves, each wave owns one 16(batch) x 16(out-col) tile.
// grid = (NOUT/16, BATCH/(8*16)) = (8,4) -> 256 waves, 150 WMMAs each.
__global__ __launch_bounds__(256) void k_gemm(const _Float16* __restrict__ A,
                                              const _Float16* __restrict__ WT,
                                              const unsigned* __restrict__ maxbits,
                                              float* __restrict__ out) {
    const int lane = threadIdx.x;               // 0..31, EXEC all-ones
    const int wave = threadIdx.y;               // 0..7
    const int n0 = blockIdx.x * 16;             // output-column tile
    const int m0 = (blockIdx.y * 8 + wave) * 16;// batch-row tile
    const int lo = lane & 15;
    const int hi = lane >> 4;

    // A fragment: lane -> row M = lo; K = hi*16 + h (contig)
    const _Float16* arow = A + (size_t)(m0 + lo) * KP3;
    // B fragment from transposed W: lane -> col N = lo; K = hi*16 + h (contig)
    const int jpos = 2 * (n0 + lo);
    const _Float16* brow_p = WT + (size_t)jpos * KP3;
    const _Float16* brow_n = WT + (size_t)(jpos + 1) * KP3;

    v8f cpos = {};
    v8f cneg = {};
    for (int kb = 0; kb < KP3; kb += 32) {
        v8h alo = *(const v8h*)(arow + kb + hi * 16);
        v8h ahi = *(const v8h*)(arow + kb + hi * 16 + 8);
        v8h bp0 = *(const v8h*)(brow_p + kb + hi * 16);
        v8h bp1 = *(const v8h*)(brow_p + kb + hi * 16 + 8);
        v8h bn0 = *(const v8h*)(brow_n + kb + hi * 16);
        v8h bn1 = *(const v8h*)(brow_n + kb + hi * 16 + 8);
        v16h a, bp, bn;
#pragma unroll
        for (int t = 0; t < 8; ++t) {
            a[t] = alo[t];  a[8 + t] = ahi[t];
            bp[t] = bp0[t]; bp[8 + t] = bp1[t];
            bn[t] = bn0[t]; bn[8 + t] = bn1[t];
        }
        cpos = __builtin_amdgcn_wmma_f32_16x16x32_f16(
            false, a, false, bp, (short)0, cpos, false, false);
        cneg = __builtin_amdgcn_wmma_f32_16x16x32_f16(
            false, a, false, bn, (short)0, cneg, false, false);
    }

    // out = (I_pos - I_neg) / k_I,  k_I = K_V * (G_MAX-G_MIN)/max_w
    const float maxw = __uint_as_float(*maxbits);
    const float inv_kI = maxw / (K_V * (G_MAX - G_MIN));
#pragma unroll
    for (int r = 0; r < 8; ++r) {
        const int row = m0 + r + hi * 8;        // C layout: M = r + hi*8
        out[(size_t)row * NOUT + n0 + lo] = (cpos[r] - cneg[r]) * inv_kI;
    }
}

extern "C" void kernel_launch(void* const* d_in, const int* in_sizes, int n_in,
                              void* d_out, int out_size, void* d_ws, size_t ws_size,
                              hipStream_t stream) {
    const float* x  = (const float*)d_in[0];   // (512, 784)
    const float* wp = (const float*)d_in[1];   // (784, 128)
    const float* wn = (const float*)d_in[2];   // (784, 128)
    const float* bp = (const float*)d_in[3];   // (128,)
    const float* bn = (const float*)d_in[4];   // (128,)
    float* out = (float*)d_out;                // (512, 128)

    unsigned char* ws = (unsigned char*)d_ws;
    unsigned* maxbits = (unsigned*)ws;
    _Float16* A  = (_Float16*)(ws + 256);
    _Float16* WT = (_Float16*)(ws + 256 + (size_t)BATCH * KP3 * sizeof(_Float16));

    k_init<<<1, 1, 0, stream>>>(maxbits);
    k_maxw<<<128, 256, 0, stream>>>(wp, wn, maxbits);
    k_build_wt<<<dim3((KP + 255) / 256, NCOL), 256, 0, stream>>>(wp, wn, bp, bn,
                                                                 maxbits, WT);
    k_build_a<<<dim3((KP + 255) / 256, BATCH), 256, 0, stream>>>(x, A);
    k_gemm<<<dim3(NOUT / 16, BATCH / 128), dim3(32, 8), 0, stream>>>(A, WT,
                                                                     maxbits, out);
}
